// W8A16LinearLayer_1623497638588
// MI455X (gfx1250) — matrix operand order repaired and checked by reading, not compiled, not run on hardware
//
#include <hip/hip_runtime.h>

typedef __attribute__((ext_vector_type(16))) _Float16 v16h;
typedef __attribute__((ext_vector_type(8)))  _Float16 v8h;
typedef __attribute__((ext_vector_type(4)))  _Float16 v4h;
typedef __attribute__((ext_vector_type(2)))  __fp16   h2;   // cvt_pkrtz return type
typedef __attribute__((ext_vector_type(8)))  float    v8f;

#define MDIM 8192      // B*S = 4*2048
#define NDIM 11008     // OUT
#define KDIM 4096      // IN

union AB16 { v16h v; v8h h[2]; };
union H8  { v8h v; h2 p[4]; };

// ---------------------------------------------------------------------------
// CDNA5 async global->LDS copy (ASYNCcnt-tracked, no VGPR round trip).
// LDS destination address taken from the vdst VGPR; per-lane global address in VGPR pair.
// ---------------------------------------------------------------------------
__device__ __forceinline__ void async_copy_b128(unsigned ldsOff, const void* gptr) {
    asm volatile("global_load_async_to_lds_b128 %0, %1, off"
                 :: "v"(ldsOff), "v"((unsigned long long)(uintptr_t)gptr)
                 : "memory");
}
__device__ __forceinline__ void wait_async0() {
    asm volatile("s_wait_asynccnt 0" ::: "memory");
}

// ---------------------------------------------------------------------------
// Prep kernel 1: dequantize W (int32, 0..255) -> f16, once per call.
// ---------------------------------------------------------------------------
__global__ __launch_bounds__(256)
void dequant_w(const int* __restrict__ W, const float* __restrict__ scales,
               const int* __restrict__ zps, _Float16* __restrict__ Wh)
{
    const size_t i8 = ((size_t)blockIdx.x * 256 + threadIdx.x) * 8;
    const int n = (int)(i8 >> 12);           // row = i8 / KDIM
    const float s = scales[n];
    const float z = (float)zps[n];
    const int4 w0 = *(const int4*)(W + i8);
    const int4 w1 = *(const int4*)(W + i8 + 4);
    H8 u;
    u.p[0] = __builtin_amdgcn_cvt_pkrtz(((float)w0.x - z) * s, ((float)w0.y - z) * s);
    u.p[1] = __builtin_amdgcn_cvt_pkrtz(((float)w0.z - z) * s, ((float)w0.w - z) * s);
    u.p[2] = __builtin_amdgcn_cvt_pkrtz(((float)w1.x - z) * s, ((float)w1.y - z) * s);
    u.p[3] = __builtin_amdgcn_cvt_pkrtz(((float)w1.z - z) * s, ((float)w1.w - z) * s);
    *(v8h*)(Wh + i8) = u.v;
}

// ---------------------------------------------------------------------------
// Prep kernel 2: activations f32 -> f16, once per call.
// ---------------------------------------------------------------------------
__global__ __launch_bounds__(256)
void cvt_a(const float* __restrict__ A, _Float16* __restrict__ Ah)
{
    const size_t i8 = ((size_t)blockIdx.x * 256 + threadIdx.x) * 8;
    const float4 a0 = *(const float4*)(A + i8);
    const float4 a1 = *(const float4*)(A + i8 + 4);
    H8 u;
    u.p[0] = __builtin_amdgcn_cvt_pkrtz(a0.x, a0.y);
    u.p[1] = __builtin_amdgcn_cvt_pkrtz(a0.z, a0.w);
    u.p[2] = __builtin_amdgcn_cvt_pkrtz(a1.x, a1.y);
    u.p[3] = __builtin_amdgcn_cvt_pkrtz(a1.z, a1.w);
    *(v8h*)(Ah + i8) = u.v;
}

// ---------------------------------------------------------------------------
// Main GEMM: f16 x f16 -> f32, 128(M) x 256(N) x 64(K) block tile, 8 wave32s
// in a 2x4 grid, each wave owns a 64x64 C sub-tile = 4x4 WMMA accumulators.
// Double-buffered LDS; staging via async global->LDS (overlapped with WMMA).
// ---------------------------------------------------------------------------
#define BM   128
#define BN   256
#define BK   64
#define LDST 72        // padded LDS stride in halves (16 distinct bank groups)

__global__ __launch_bounds__(256)
void w8a16_gemm_f16(const _Float16* __restrict__ Ah,
                    const _Float16* __restrict__ Wh,
                    const float* __restrict__ bias,
                    float* __restrict__ C)
{
    __shared__ _Float16 sA[2][BM * LDST];   // 2 x 18 KB
    __shared__ _Float16 sB[2][BN * LDST];   // 2 x 37 KB

    const int tid  = threadIdx.x;
    const int lane = tid & 31;
    const int wave = tid >> 5;
    const int wm   = wave >> 2;          // 0..1 : 64-row strip
    const int wn   = wave & 3;           // 0..3 : 64-col strip

    const int bm = blockIdx.y * BM;
    const int bn = blockIdx.x * BN;

    const _Float16* aBase = Ah + (size_t)bm * KDIM;
    const _Float16* bBase = Wh + (size_t)bn * KDIM;

    // Issue per-lane async b128 copies for one K-tile into buffer `buf`.
    auto stage = [&](int ktN, int buf) {
        #pragma unroll
        for (int i = 0; i < 4; ++i) {            // A: 128x64 halves
            const int c   = tid + i * 256;
            const int r   = c >> 3;
            const int col = (c & 7) << 3;
            async_copy_b128((unsigned)(uintptr_t)&sA[buf][r * LDST + col],
                            aBase + (size_t)r * KDIM + ktN + col);
        }
        #pragma unroll
        for (int i = 0; i < 8; ++i) {            // B: 256x64 halves
            const int c   = tid + i * 256;
            const int r   = c >> 3;
            const int col = (c & 7) << 3;
            async_copy_b128((unsigned)(uintptr_t)&sB[buf][r * LDST + col],
                            bBase + (size_t)r * KDIM + ktN + col);
        }
    };

    v8f zero = {};
    v8f acc[4][4];
    #pragma unroll
    for (int im = 0; im < 4; ++im)
        #pragma unroll
        for (int jn = 0; jn < 4; ++jn)
            acc[im][jn] = zero;

    // Prologue: stage tile 0 into buffer 0.
    stage(0, 0);
    wait_async0();
    __syncthreads();

    for (int kt = 0; kt < KDIM; kt += BK) {
        const int cur = (kt >> 6) & 1;

        // Prefetch next K-tile into the other buffer (overlaps with WMMAs).
        if (kt + BK < KDIM)
            stage(kt + BK, cur ^ 1);

        // ---- two 16x16x32 WMMA k-steps: 16 ds_load_b128 : 16 wmma each ----
        #pragma unroll
        for (int ks = 0; ks < BK; ks += 32) {
            AB16 a[4], b[4];
            // A frag: lanes 0-15 K-chunks {0..7,16..23}; lanes 16-31 {8..15,24..31}
            const int ak = ks + ((lane & 16) ? 8 : 0);
            #pragma unroll
            for (int im = 0; im < 4; ++im) {
                const int ar = wm * 64 + im * 16 + (lane & 15);
                a[im].h[0] = *(const v8h*)(&sA[cur][ar * LDST + ak]);
                a[im].h[1] = *(const v8h*)(&sA[cur][ar * LDST + ak + 16]);
            }
            // B frag: lane holds col N=lane&15; lanes 0-15 K-chunks {0..7,16..23}; lanes 16-31 {8..15,24..31}
            const int bk = ks + ((lane & 16) ? 8 : 0);
            #pragma unroll
            for (int jn = 0; jn < 4; ++jn) {
                const int br = wn * 64 + jn * 16 + (lane & 15);
                b[jn].h[0] = *(const v8h*)(&sB[cur][br * LDST + bk]);
                b[jn].h[1] = *(const v8h*)(&sB[cur][br * LDST + bk + 16]);
            }
            #pragma unroll
            for (int im = 0; im < 4; ++im)
                #pragma unroll
                for (int jn = 0; jn < 4; ++jn)
                    acc[im][jn] = __builtin_amdgcn_wmma_f32_16x16x32_f16(
                        false, a[im].v, false, b[jn].v,
                        (short)0, acc[im][jn], false, false);
        }

        // All our async loads complete, then block-wide rendezvous: next
        // iteration may read buf^1 and overwrite buf (reads of buf done).
        wait_async0();
        __syncthreads();
    }

    // ---- epilogue: + bias, store f32 ----
    #pragma unroll
    for (int jn = 0; jn < 4; ++jn) {
        const int n   = bn + wn * 64 + jn * 16 + (lane & 15);
        const float bv = bias[n];
        #pragma unroll
        for (int im = 0; im < 4; ++im) {
            const int m0 = bm + wm * 64 + im * 16 + ((lane & 16) ? 8 : 0);
            #pragma unroll
            for (int j = 0; j < 8; ++j)
                C[(size_t)(m0 + j) * NDIM + n] = acc[im][jn][j] + bv;
        }
    }
}

// ---------------------------------------------------------------------------
// Fallback (zero-workspace) fused kernel: 128x128x64 tiles, in-loop dequant.
// ---------------------------------------------------------------------------
__global__ __launch_bounds__(256)
void w8a16_gemm_fused(const float* __restrict__ A,
                      const int*   __restrict__ W,
                      const float* __restrict__ scales,
                      const int*   __restrict__ zps,
                      const float* __restrict__ bias,
                      float*       __restrict__ C)
{
    __shared__ _Float16 sA[128 * LDST];
    __shared__ _Float16 sW[128 * LDST];

    const int tid  = threadIdx.x;
    const int lane = tid & 31;
    const int wave = tid >> 5;
    const int wm   = wave >> 2;
    const int wn   = wave & 3;

    const int bm = blockIdx.y * 128;
    const int bn = blockIdx.x * 128;

    int   rowA[8], colA[8];
    float wsc[8];
    int   wzp[8];
    #pragma unroll
    for (int i = 0; i < 8; ++i) {
        int idx  = tid + i * 256;
        rowA[i]  = idx >> 4;
        colA[i]  = (idx & 15) << 2;
        int n    = bn + rowA[i];
        wsc[i]   = scales[n];
        wzp[i]   = zps[n];
    }

    v8f zero = {};
    v8f acc[4][2];
    #pragma unroll
    for (int im = 0; im < 4; ++im)
        #pragma unroll
        for (int jn = 0; jn < 2; ++jn)
            acc[im][jn] = zero;

    for (int kt = 0; kt < KDIM; kt += 64) {
        __syncthreads();
        #pragma unroll
        for (int i = 0; i < 8; ++i) {
            const float4 v = *(const float4*)(A + (size_t)(bm + rowA[i]) * KDIM + kt + colA[i]);
            v4h h;
            h.x = (_Float16)v.x; h.y = (_Float16)v.y;
            h.z = (_Float16)v.z; h.w = (_Float16)v.w;
            *(v4h*)(&sA[rowA[i] * LDST + colA[i]]) = h;
        }
        #pragma unroll
        for (int i = 0; i < 8; ++i) {
            const int4 w = *(const int4*)(W + (size_t)(bn + rowA[i]) * KDIM + kt + colA[i]);
            const float s = wsc[i];
            const float z = (float)wzp[i];
            v4h h;
            h.x = (_Float16)(((float)w.x - z) * s);
            h.y = (_Float16)(((float)w.y - z) * s);
            h.z = (_Float16)(((float)w.z - z) * s);
            h.w = (_Float16)(((float)w.w - z) * s);
            *(v4h*)(&sW[rowA[i] * LDST + colA[i]]) = h;
        }
        __syncthreads();

        #pragma unroll
        for (int ks = 0; ks < 64; ks += 32) {
            AB16 a[4], b[2];
            const int ak = ks + ((lane & 16) ? 8 : 0);
            #pragma unroll
            for (int im = 0; im < 4; ++im) {
                const int ar = wm * 64 + im * 16 + (lane & 15);
                a[im].h[0] = *(const v8h*)(&sA[ar * LDST + ak]);
                a[im].h[1] = *(const v8h*)(&sA[ar * LDST + ak + 16]);
            }
            const int bk = ks + ((lane & 16) ? 8 : 0);
            #pragma unroll
            for (int jn = 0; jn < 2; ++jn) {
                const int br = wn * 32 + jn * 16 + (lane & 15);
                b[jn].h[0] = *(const v8h*)(&sW[br * LDST + bk]);
                b[jn].h[1] = *(const v8h*)(&sW[br * LDST + bk + 16]);
            }
            #pragma unroll
            for (int im = 0; im < 4; ++im)
                #pragma unroll
                for (int jn = 0; jn < 2; ++jn)
                    acc[im][jn] = __builtin_amdgcn_wmma_f32_16x16x32_f16(
                        false, a[im].v, false, b[jn].v,
                        (short)0, acc[im][jn], false, false);
        }
    }

    #pragma unroll
    for (int jn = 0; jn < 2; ++jn) {
        const int n  = bn + wn * 32 + jn * 16 + (lane & 15);
        const float bv = bias[n];
        #pragma unroll
        for (int im = 0; im < 4; ++im) {
            const int m0 = bm + wm * 64 + im * 16 + ((lane & 16) ? 8 : 0);
            #pragma unroll
            for (int j = 0; j < 8; ++j)
                C[(size_t)(m0 + j) * NDIM + n] = acc[im][jn][j] + bv;
        }
    }
}

// ---------------------------------------------------------------------------
extern "C" void kernel_launch(void* const* d_in, const int* in_sizes, int n_in,
                              void* d_out, int out_size, void* d_ws, size_t ws_size,
                              hipStream_t stream) {
    const float* input = (const float*)d_in[0];   // [4,2048,4096] f32
    const int*   w     = (const int*)d_in[1];     // [11008,4096] int32
    const float* sc    = (const float*)d_in[2];   // [11008] f32
    const int*   zp    = (const int*)d_in[3];     // [11008] int32
    const float* bias  = (const float*)d_in[4];   // [1,11008] f32
    float* out = (float*)d_out;                   // [4,2048,11008] f32

    const size_t needW = (size_t)NDIM * KDIM * sizeof(_Float16);  // 90.2 MB
    const size_t needA = (size_t)MDIM * KDIM * sizeof(_Float16);  // 67.1 MB

    if (ws_size >= needW + needA) {
        _Float16* Wh = (_Float16*)d_ws;
        _Float16* Ah = (_Float16*)((char*)d_ws + needW);
        dequant_w<<<(int)(((size_t)NDIM * KDIM) / (256 * 8)), 256, 0, stream>>>(w, sc, zp, Wh);
        cvt_a<<<(int)(((size_t)MDIM * KDIM) / (256 * 8)), 256, 0, stream>>>(input, Ah);
        dim3 grid(NDIM / BN, MDIM / BM);          // (43, 64)
        w8a16_gemm_f16<<<grid, dim3(256), 0, stream>>>(Ah, Wh, bias, out);
    } else {
        dim3 grid(NDIM / 128, MDIM / 128);        // (86, 64)
        w8a16_gemm_fused<<<grid, dim3(256), 0, stream>>>(input, w, sc, zp, bias, out);
    }
}
